// PhysicsInformedEncoder_88184268521779
// MI455X (gfx1250) — compile-verified, matrix operand order checked by reading, not run on hardware
//
#include <hip/hip_runtime.h>
#include <cmath>

typedef _Float16 f16;
typedef __attribute__((ext_vector_type(16))) _Float16 v16h;
typedef __attribute__((ext_vector_type(8)))  _Float16 h8;
typedef __attribute__((ext_vector_type(8)))  float    v8f;
typedef __attribute__((ext_vector_type(4)))  int      v4i;

#define BSZ  1024
#define SEQ  65
#define DM   256
#define NH   8
#define DH   32
#define NL   4
#define HID  512
#define ROWS (BSZ*SEQ)     // 66560
#define TDIM (SEQ*DM)      // 16640
#define IMGN 64
#define PIX  (IMGN*IMGN)   // 4096

// Async global->LDS staging (CDNA5 GLOBAL_LOAD_ASYNC_TO_LDS_B128, ASYNCcnt),
// guarded so the file still compiles if the builtins are absent.
#if defined(__has_builtin)
# if __has_builtin(__builtin_amdgcn_global_load_async_to_lds_b128)
#  define HAVE_ASYNC_LDS 1
# endif
# if __has_builtin(__builtin_amdgcn_s_wait_asynccnt)
#  define HAVE_WAIT_ASYNC 1
# endif
#endif

#define AS1 __attribute__((address_space(1)))
#define AS3 __attribute__((address_space(3)))

// ---------------------------------------------------------------------------
// Fragment loader for v_wmma_f32_16x16x32_f16.
// ISA 16-bit A-matrix 16x32 layout: lane l (lh=l/16, lr=l%16) holds row lr
// with k-elements {k0..k0+7, k0+16..k0+23}, k0 = kstep + lh*8.
// B pre-transposed to [N,K] uses the identical pattern.
// ---------------------------------------------------------------------------
__device__ inline v16h load_frag(const f16* __restrict__ row, int k0) {
    h8 lo = *(const h8*)(row + k0);
    h8 hi = *(const h8*)(row + k0 + 16);
    return __builtin_shufflevector(lo, hi, 0,1,2,3,4,5,6,7,8,9,10,11,12,13,14,15);
}

__device__ inline v8f wmma_f16(v16h a, v16h b, v8f c) {
    return __builtin_amdgcn_wmma_f32_16x16x32_f16(false, a, false, b,
                                                  (short)0, c, false, false);
}

// ---------------------------------------------------------------------------
// WMMA GEMM: C[M,N] = A[M,K](f16) * Bt[N,K](f16)^T + bias
// Block = (32,8): 8 waves. Block tile = 256 rows x 64 cols; each wave owns
// 32 rows x 64 cols (2 A-frags x 4 B-frags = 8 WMMA per k-step).
// Bt panel (64x32 per k-step) is staged into LDS once per block, double
// buffered via async global->LDS copies when available.
// EPI: 0 = f16 out, 1 = f32 accumulate (residual), 2 = GELU -> f16, 3 = f32 out
// Requires: M % 256 == 0, N % 64 == 0, K % 32 == 0.
// ---------------------------------------------------------------------------
template<int EPI>
__global__ __launch_bounds__(256)
void gemm_wmma(const f16* __restrict__ A, const f16* __restrict__ Bt,
               const float* __restrict__ bias, int M, int N, int K,
               f16* __restrict__ Oh, float* __restrict__ Of)
{
    __shared__ __align__(16) f16 bl[2][64 * 32];   // double-buffered B panel

    const int lane = threadIdx.x;
    const int wave = threadIdx.y;
    const int tid  = wave * 32 + lane;
    const int lh = lane >> 4, lr = lane & 15;
    const int m0 = blockIdx.x * 256 + wave * 32;
    const int n0 = blockIdx.y * 64;

    // Panel staging: 256 threads x 16B cover the 64x32 f16 chunk.
    const int srow = tid >> 2;              // 0..63
    const int scol = (tid & 3) * 8;         // 0,8,16,24
    const f16* sg = Bt + (size_t)(n0 + srow) * K + scol;
    f16* const sl[2] = { &bl[0][srow * 32 + scol], &bl[1][srow * 32 + scol] };

    auto stage = [&](int ks, int buf) {
#if defined(HAVE_ASYNC_LDS) && defined(HAVE_WAIT_ASYNC)
        __builtin_amdgcn_global_load_async_to_lds_b128(
            (AS1 v4i*)(void*)(sg + ks), (AS3 v4i*)(void*)sl[buf], 0, 0);
#else
        *(h8*)sl[buf] = *(const h8*)(sg + ks);
#endif
    };

    v8f acc[2][4] = {};
    const f16* arow0 = A + (size_t)(m0 + lr) * K;
    const f16* arow1 = A + (size_t)(m0 + 16 + lr) * K;

    const int nk = K / 32;
    stage(0, 0);
    int cur = 0;
    for (int ki = 0; ki < nk; ki++) {
#if defined(HAVE_ASYNC_LDS) && defined(HAVE_WAIT_ASYNC)
        __builtin_amdgcn_s_wait_asynccnt(0);
#endif
        __syncthreads();                    // panel[cur] visible to all waves
        if (ki + 1 < nk) stage((ki + 1) * 32, cur ^ 1);

        const int k0 = ki * 32 + lh * 8;
        __builtin_prefetch(arow0 + ki * 32 + 256, 0, 1);  // global_prefetch
        v16h a0 = load_frag(arow0, k0);
        v16h a1 = load_frag(arow1, k0);
        const f16* lb = &bl[cur][0];
#pragma unroll
        for (int t = 0; t < 4; t++) {
            v16h b = load_frag(lb + (size_t)(t * 16 + lr) * 32, lh * 8);
            acc[0][t] = wmma_f16(a0, b, acc[0][t]);
            acc[1][t] = wmma_f16(a1, b, acc[1][t]);
        }
        cur ^= 1;
    }

#pragma unroll
    for (int mi = 0; mi < 2; mi++) {
#pragma unroll
        for (int t = 0; t < 4; t++) {
            const int n = n0 + t * 16 + lr;
            const float bv = bias ? bias[n] : 0.0f;
#pragma unroll
            for (int j = 0; j < 8; j++) {
                const int m = m0 + mi * 16 + j + 8 * lh;  // C layout
                float v = acc[mi][t][j] + bv;
                const size_t off = (size_t)m * N + n;
                if (EPI == 0) {
                    Oh[off] = (f16)v;
                } else if (EPI == 1) {
                    Of[off] += v;
                } else if (EPI == 2) {
                    float g = 0.5f * v * (1.0f + tanhf(0.7978845608028654f *
                                                       (v + 0.044715f * v * v * v)));
                    Oh[off] = (f16)g;
                } else {
                    Of[off] = v;
                }
            }
        }
    }
}

// ---------------------------------------------------------------------------
// LayerNorm over D=256, one row per 256-thread block; writes f16.
// ---------------------------------------------------------------------------
__global__ __launch_bounds__(256)
void ln_kernel(const float* __restrict__ x, const float* __restrict__ g,
               const float* __restrict__ b, f16* __restrict__ out)
{
    __shared__ float red[8];
    const int row = blockIdx.x;
    const int t = threadIdx.x;
    const float v = x[(size_t)row * DM + t];

    float s = v;
#pragma unroll
    for (int m = 16; m; m >>= 1) s += __shfl_xor(s, m, 32);
    if ((t & 31) == 0) red[t >> 5] = s;
    __syncthreads();
    float mean = 0.0f;
#pragma unroll
    for (int i = 0; i < 8; i++) mean += red[i];
    mean *= (1.0f / DM);
    __syncthreads();

    const float d = v - mean;
    float q = d * d;
#pragma unroll
    for (int m = 16; m; m >>= 1) q += __shfl_xor(q, m, 32);
    if ((t & 31) == 0) red[t >> 5] = q;
    __syncthreads();
    float var = 0.0f;
#pragma unroll
    for (int i = 0; i < 8; i++) var += red[i];
    var *= (1.0f / DM);

    const float r = rsqrtf(var + 1e-5f);
    out[(size_t)row * DM + t] = (f16)(d * r * g[t] + b[t]);
}

// ---------------------------------------------------------------------------
// Fused LSA attention, one wave per (b, h). S=65 padded to 80 (5 tiles),
// K padded to 96 for the attn@V pass. All matmuls via WMMA.
// ---------------------------------------------------------------------------
__global__ __launch_bounds__(32)
void attn_kernel(const f16* __restrict__ qkv, f16* __restrict__ o,
                 const float* __restrict__ temp_l)
{
    __shared__ __align__(16) f16 Vt[DH][96];   // V^T, zero padded keys
    __shared__ __align__(16) f16 At[16][96];   // softmax probs for one q-tile

    const int lane = threadIdx.x;
    const int lh = lane >> 4, lr = lane & 15;
    const int b = blockIdx.x / NH;
    const int h = blockIdx.x % NH;
    const float scale = 1.0f / temp_l[h];

    const size_t rs = 3 * DM;  // qkv row stride = 768
    const f16* qbase = qkv + (size_t)b * SEQ * rs + h * DH;
    const f16* kbase = qbase + DM;
    const f16* vbase = qbase + 2 * DM;

    for (int s = 0; s < 96; s++) {
        f16 val = (f16)0;
        if (s < SEQ) val = vbase[(size_t)s * rs + lane];
        Vt[lane][s] = val;
    }
    __syncthreads();

    for (int qt = 0; qt < 5; qt++) {
        int qr = qt * 16 + lr; if (qr >= SEQ) qr = SEQ - 1;
        const v16h aq = load_frag(qbase + (size_t)qr * rs, lh * 8);

        v8f sc[5];
#pragma unroll
        for (int kt = 0; kt < 5; kt++) {
            int kr = kt * 16 + lr; if (kr >= SEQ) kr = SEQ - 1;
            v16h bk = load_frag(kbase + (size_t)kr * rs, lh * 8);
            v8f z = {};
            sc[kt] = wmma_f16(aq, bk, z);
        }

        float rmax[8];
#pragma unroll
        for (int j = 0; j < 8; j++) {
            const int m = qt * 16 + j + 8 * lh;    // global query index
            float best = -3.0e38f;
#pragma unroll
            for (int kt = 0; kt < 5; kt++) {
                const int n = kt * 16 + lr;        // global key index
                float v = sc[kt][j] * scale;
                if (n >= SEQ || n == m) v = -1.0e9f;
                sc[kt][j] = v;
                best = fmaxf(best, v);
            }
            rmax[j] = best;
        }
#pragma unroll
        for (int j = 0; j < 8; j++)
#pragma unroll
            for (int mk = 8; mk; mk >>= 1)
                rmax[j] = fmaxf(rmax[j], __shfl_xor(rmax[j], mk, 32));

        float rsum[8];
#pragma unroll
        for (int j = 0; j < 8; j++) {
            float sum = 0.0f;
#pragma unroll
            for (int kt = 0; kt < 5; kt++) {
                float e = __expf(sc[kt][j] - rmax[j]);
                sc[kt][j] = e;
                sum += e;
            }
            rsum[j] = sum;
        }
#pragma unroll
        for (int j = 0; j < 8; j++)
#pragma unroll
            for (int mk = 8; mk; mk >>= 1)
                rsum[j] += __shfl_xor(rsum[j], mk, 32);

#pragma unroll
        for (int j = 0; j < 8; j++) {
            const float inv = 1.0f / rsum[j];
            const int rrow = j + 8 * lh;
#pragma unroll
            for (int kt = 0; kt < 5; kt++)
                At[rrow][kt * 16 + lr] = (f16)(sc[kt][j] * inv);
            At[rrow][80 + lr] = (f16)0;            // zero pad keys 80..95
        }
        __syncthreads();

#pragma unroll
        for (int nt = 0; nt < 2; nt++) {
            v8f acc = {};
#pragma unroll
            for (int ks = 0; ks < 96; ks += 32) {
                v16h aa = load_frag(&At[lr][0], ks + lh * 8);
                v16h bv = load_frag(&Vt[nt * 16 + lr][0], ks + lh * 8);
                acc = wmma_f16(aa, bv, acc);
            }
#pragma unroll
            for (int j = 0; j < 8; j++) {
                const int m = qt * 16 + j + 8 * lh;
                if (m < SEQ)
                    o[((size_t)(b * SEQ) + m) * DM + h * DH + nt * 16 + lr] = (f16)acc[j];
            }
        }
        __syncthreads();
    }
}

// ---------------------------------------------------------------------------
// Weight convert + transpose: W[K,N] f32 -> Wt[N,K] f16
// ---------------------------------------------------------------------------
__global__ void wtrans(const float* __restrict__ W, f16* __restrict__ Wt,
                       int K, int N)
{
    size_t i = (size_t)blockIdx.x * blockDim.x + threadIdx.x;
    if (i >= (size_t)K * N) return;
    int k = (int)(i / N), n = (int)(i % N);
    Wt[(size_t)n * K + k] = (f16)W[i];
}

__global__ void cvt16(const float* __restrict__ x, f16* __restrict__ y, size_t n)
{
    size_t i = (size_t)blockIdx.x * blockDim.x + threadIdx.x;
    if (i < n) y[i] = (f16)x[i];
}

// ---------------------------------------------------------------------------
// image_to_source scatter: one block per batch, 4096-bin histogram in LDS.
// ---------------------------------------------------------------------------
__global__ __launch_bounds__(256)
void scatter_kernel(const float* __restrict__ ksig, const float* __restrict__ dist,
                    const float* __restrict__ imgs, float* __restrict__ src)
{
    __shared__ float s_sum[PIX];
    __shared__ float s_cnt[PIX];
    const int b = blockIdx.x;
    for (int i = threadIdx.x; i < PIX; i += 256) { s_sum[i] = 0.0f; s_cnt[i] = 0.0f; }
    __syncthreads();

    const float pw = (3.232f - (-3.323f)) / 64.0f;
    for (int p = threadIdx.x; p < PIX; p += 256) {
        const int i = p >> 6, j = p & 63;
        const float gx = (float)(i - 31) * pw;
        const float gy = (float)(j - 31) * pw;
        float r = sqrtf(gx * gx + gy * gy);
        if (r == 0.0f) r = 1.0f;
        const float er = ksig[(size_t)b * PIX + p] * dist[(size_t)b * PIX + p];
        const float xdef = er * gx / r;
        const float ydef = er * gy / r;
        float fx = (gx - xdef) / pw + 32.0f;
        float fy = (gy - ydef) / pw + 32.0f;
        fx = fminf(fmaxf(fx, 0.0f), 63.0f);
        fy = fminf(fmaxf(fy, 0.0f), 63.0f);
        const int idx = ((int)fx) * 64 + (int)fy;
        atomicAdd(&s_sum[idx], imgs[(size_t)b * PIX + p]);
        atomicAdd(&s_cnt[idx], 1.0f);
    }
    __syncthreads();
    for (int p = threadIdx.x; p < PIX; p += 256) {
        const float s = s_sum[p], c = s_cnt[p];
        src[(size_t)b * PIX + p] = (c != 0.0f) ? s / c : s;
    }
}

// ---------------------------------------------------------------------------
extern "C" void kernel_launch(void* const* d_in, const int* in_sizes, int n_in,
                              void* d_out, int out_size, void* d_ws, size_t ws_size,
                              hipStream_t stream)
{
    (void)in_sizes; (void)n_in; (void)out_size; (void)ws_size;

    const float* patches = (const float*)d_in[0];
    const float* images  = (const float*)d_in[1];
    const float* dist    = (const float*)d_in[2];
    const float* ln1s    = (const float*)d_in[3];
    const float* ln1b    = (const float*)d_in[4];
    const float* Wqkv    = (const float*)d_in[5];
    const float* bqkv    = (const float*)d_in[6];
    const float* Wo      = (const float*)d_in[7];
    const float* bo      = (const float*)d_in[8];
    const float* temp    = (const float*)d_in[9];
    const float* ln2s    = (const float*)d_in[10];
    const float* ln2b    = (const float*)d_in[11];
    const float* W1      = (const float*)d_in[12];
    const float* b1      = (const float*)d_in[13];
    const float* W2      = (const float*)d_in[14];
    const float* b2      = (const float*)d_in[15];
    const float* Wout    = (const float*)d_in[16];
    const float* bout    = (const float*)d_in[17];

    float* out_ksig = (float*)d_out;
    float* out_src  = out_ksig + (size_t)BSZ * PIX;

    char* w = (char*)d_ws;
    auto alloc = [&](size_t bytes) {
        char* p = w;
        w += (bytes + 255) & ~(size_t)255;
        return p;
    };
    float* x   = (float*)alloc((size_t)ROWS * DM * 4);
    f16* hbuf  = (f16*)alloc((size_t)ROWS * DM * 2);        // also xh [B,16640] at end
    f16* qkvb  = (f16*)alloc((size_t)ROWS * 3 * DM * 2);
    f16* obuf  = (f16*)alloc((size_t)ROWS * DM * 2);
    f16* midb  = (f16*)alloc((size_t)ROWS * HID * 2);
    f16* WqkvT = (f16*)alloc((size_t)NL * 3 * DM * DM * 2);
    f16* WoT   = (f16*)alloc((size_t)NL * DM * DM * 2);
    f16* W1T   = (f16*)alloc((size_t)NL * HID * DM * 2);
    f16* W2T   = (f16*)alloc((size_t)NL * DM * HID * 2);
    f16* WoutT = (f16*)alloc((size_t)PIX * TDIM * 2);

    auto launch_wt = [&](const float* W_, f16* Wt_, int K, int N) {
        size_t tot = (size_t)K * N;
        wtrans<<<dim3((unsigned)((tot + 255) / 256)), dim3(256), 0, stream>>>(W_, Wt_, K, N);
    };
    for (int l = 0; l < NL; l++) {
        launch_wt(Wqkv + (size_t)l * DM * 3 * DM, WqkvT + (size_t)l * 3 * DM * DM, DM, 3 * DM);
        launch_wt(Wo   + (size_t)l * DM * DM,     WoT   + (size_t)l * DM * DM,     DM, DM);
        launch_wt(W1   + (size_t)l * DM * HID,    W1T   + (size_t)l * HID * DM,    DM, HID);
        launch_wt(W2   + (size_t)l * HID * DM,    W2T   + (size_t)l * DM * HID,    HID, DM);
    }
    launch_wt(Wout, WoutT, TDIM, PIX);

    (void)hipMemcpyAsync(x, patches, (size_t)ROWS * DM * 4,
                         hipMemcpyDeviceToDevice, stream);

    const dim3 wblk(32, 8);
    for (int l = 0; l < NL; l++) {
        ln_kernel<<<ROWS, 256, 0, stream>>>(x, ln1s + l * DM, ln1b + l * DM, hbuf);
        gemm_wmma<0><<<dim3(ROWS / 256, (3 * DM) / 64), wblk, 0, stream>>>(
            hbuf, WqkvT + (size_t)l * 3 * DM * DM, bqkv + (size_t)l * 3 * DM,
            ROWS, 3 * DM, DM, qkvb, nullptr);
        attn_kernel<<<BSZ * NH, 32, 0, stream>>>(qkvb, obuf, temp + l * NH);
        gemm_wmma<1><<<dim3(ROWS / 256, DM / 64), wblk, 0, stream>>>(
            obuf, WoT + (size_t)l * DM * DM, bo + l * DM,
            ROWS, DM, DM, nullptr, x);
        ln_kernel<<<ROWS, 256, 0, stream>>>(x, ln2s + l * DM, ln2b + l * DM, hbuf);
        gemm_wmma<2><<<dim3(ROWS / 256, HID / 64), wblk, 0, stream>>>(
            hbuf, W1T + (size_t)l * HID * DM, b1 + (size_t)l * HID,
            ROWS, HID, DM, midb, nullptr);
        gemm_wmma<1><<<dim3(ROWS / 256, DM / 64), wblk, 0, stream>>>(
            midb, W2T + (size_t)l * DM * HID, b2 + l * DM,
            ROWS, DM, HID, nullptr, x);
    }

    cvt16<<<(unsigned)(((size_t)ROWS * DM + 255) / 256), 256, 0, stream>>>(
        x, hbuf, (size_t)ROWS * DM);
    gemm_wmma<3><<<dim3(BSZ / 256, PIX / 64), wblk, 0, stream>>>(
        hbuf, WoutT, bout, BSZ, PIX, TDIM, nullptr, out_ksig);

    scatter_kernel<<<BSZ, 256, 0, stream>>>(out_ksig, dist, images, out_src);
}
